// MultiHeadedAttention_54597624266868
// MI455X (gfx1250) — matrix operand order repaired and checked by reading, not compiled, not run on hardware
//
#include <hip/hip_runtime.h>
#include <hip/hip_bf16.h>
#include <math.h>
#include <stdint.h>

typedef __attribute__((ext_vector_type(16))) _Float16 v16h;
typedef __attribute__((ext_vector_type(8)))  _Float16 v8h;
typedef __attribute__((ext_vector_type(8)))  float    v8f;
typedef unsigned int v4u __attribute__((ext_vector_type(4)));
typedef int          v8i __attribute__((ext_vector_type(8)));
typedef int          v4i __attribute__((ext_vector_type(4)));

#define DEV __device__ __forceinline__

constexpr int B_  = 8;
constexpr int S_  = 2048;
constexpr int E_  = 256;
constexpr int H_  = 8;
constexpr int HD_ = 32;
constexpr int BS_ = B_ * S_;
constexpr float INV_SQRT_E = 0.0625f;   // 1/sqrt(256)
constexpr float MASK_EPS   = 1e-9f;

// ---------------------------------------------------------------------------
// WMMA fragment loaders (wave32), contiguous k order (k = 16*half + s), same for A and B.
// A fragment 16x32 f16: lane<16 -> M=lane, K=0..15;
//                       lane>=16 -> M=lane-16, K=16..31
DEV v16h load_a_frag(const _Float16* __restrict__ src, int ld, int mBase, int kBase, int lane) {
    const int mr  = lane & 15;
    const int off = (lane >> 4) * 16;
    const _Float16* p = src + (size_t)(mBase + mr) * ld + kBase + off;
    v8h lo = *(const v8h*)(p);
    v8h hi = *(const v8h*)(p + 8);
    return __builtin_shufflevector(lo, hi, 0,1,2,3,4,5,6,7,8,9,10,11,12,13,14,15);
}

// B fragment 32x16 f16 (KxN), B[k][n] = src[nBase+n][kBase+k] (src rows are N,
// contiguous in K): lane<16 -> N=lane, K=0..15; lane>=16 -> N=lane-16, K=16..31
DEV v16h load_b_frag(const _Float16* __restrict__ src, int ld, int nBase, int kBase, int lane) {
    const int nr  = lane & 15;
    const int off = (lane >> 4) * 16;
    return *(const v16h*)(src + (size_t)(nBase + nr) * ld + kBase + off);
}

DEV v8f wmma_f16(v16h a, v16h b, v8f c) {
    return __builtin_amdgcn_wmma_f32_16x16x32_f16(false, a, false, b, (short)0, c, false, false);
}

// ---------------------------------------------------------------------------
// Tensor Data Mover staging: one D#-driven DMA of a 2D f16 tile into LDS.
// Descriptor packing per CDNA5 ISA §8.3/§8.4 (group0: count/lds/global/type,
// group1: data_size, tensor dims, tile dims, dim0 stride). This toolchain's
// builtin takes 6 args (g0 u32x4, g1 i32x8, g2 i32x4, g3 i32x4, extra i32x8, cpol).
DEV void put_bits(uint64_t* g, int lo, int width, uint64_t val) {
    val &= (width >= 64) ? ~0ull : ((1ull << width) - 1ull);
    const int w = lo >> 6, s = lo & 63;
    g[w] |= val << s;
    if (s + width > 64) g[w + 1] |= val >> (64 - s);
}

DEV void stage_tile_f16(int wid, int lane, _Float16* lds_dst, const _Float16* gsrc,
                        uint32_t tile_w, uint32_t tile_h, uint32_t row_stride_elems) {
    if (wid != 0) return;   // one TDM issue per block (EXEC is ignored by TDM)
#if __has_builtin(__builtin_amdgcn_tensor_load_to_lds)
    const uint64_t ga  = (uint64_t)(uintptr_t)gsrc;
    const uint32_t lds = (uint32_t)(uintptr_t)lds_dst;   // low 32 bits = LDS byte offset

    v4u g0;
    g0[0] = 1u;                                              // count=1, no gather
    g0[1] = lds;                                             // lds_addr
    g0[2] = (uint32_t)ga;                                    // global_addr[31:0]
    g0[3] = (uint32_t)((ga >> 32) & 0x1FFFFFFu) | (2u << 30);// global_addr[56:32] | type=2

    uint64_t q[4] = {0, 0, 0, 0};
    q[0] = (uint64_t)1u << 16;                 // workgroup_mask=0, data_size=1 (2 bytes)
    put_bits(q,  48, 32, 1u << 20);            // tensor_dim0 (ample; tile is in-bounds)
    put_bits(q,  80, 32, 1u << 20);            // tensor_dim1 (ample)
    put_bits(q, 112, 16, tile_w);              // tile_dim0 (elements)
    put_bits(q, 128, 16, tile_h);              // tile_dim1 (rows)
    put_bits(q, 160, 48, row_stride_elems);    // tensor_dim0_stride (elements)

    v8i g1;
    g1[0] = (int)(uint32_t)q[0]; g1[1] = (int)(q[0] >> 32);
    g1[2] = (int)(uint32_t)q[1]; g1[3] = (int)(q[1] >> 32);
    g1[4] = (int)(uint32_t)q[2]; g1[5] = (int)(q[2] >> 32);
    g1[6] = (int)(uint32_t)q[3]; g1[7] = (int)(q[3] >> 32);

    v4i z4 = {0, 0, 0, 0};
    v8i z8 = {0, 0, 0, 0, 0, 0, 0, 0};
    __builtin_amdgcn_tensor_load_to_lds(g0, g1, z4, z4, z8, 0);
#else
    for (uint32_t r = 0; r < tile_h; ++r)
        for (uint32_t c = (uint32_t)lane; c < tile_w; c += 32)
            lds_dst[r * tile_w + c] = gsrc[(size_t)r * row_stride_elems + c];
#endif
}

// ---------------------------------------------------------------------------
__global__ void cvt_f32_to_f16(const float* __restrict__ src, _Float16* __restrict__ dst, int n) {
    int i = blockIdx.x * blockDim.x + threadIdx.x;
    if (i < n) dst[i] = (_Float16)src[i];
}

// ---------------------------------------------------------------------------
// Fused QKV projection: out = X @ W.T + bias.
// blockIdx.y: 0=Q, 1=K (stored [B,H,S,HD] f16), 2=V (stored transposed [B,H,HD,S] f16)
__global__ __launch_bounds__(128)
void qkv_gemm_kernel(const _Float16* __restrict__ Xh,
                     const _Float16* __restrict__ Wqh,
                     const _Float16* __restrict__ Wkh,
                     const _Float16* __restrict__ Wvh,
                     const float* __restrict__ bq,
                     const float* __restrict__ bk,
                     const float* __restrict__ bvv,
                     _Float16* __restrict__ Qh,
                     _Float16* __restrict__ Kh,
                     _Float16* __restrict__ Vt) {
    const int lane   = threadIdx.x & 31;
    const int tile   = blockIdx.x * 4 + (threadIdx.x >> 5);   // 16384 tiles
    const int nBlock = tile & 15;                             // E/16 = 16
    const int mBlock = tile >> 4;                             // BS/16 = 1024
    const int mBase  = mBlock * 16;
    const int nBase  = nBlock * 16;
    const int which  = blockIdx.y;

    const _Float16* W    = (which == 0) ? Wqh : (which == 1) ? Wkh : Wvh;
    const float*    bias = (which == 0) ? bq  : (which == 1) ? bk  : bvv;

    v8f acc = {0,0,0,0,0,0,0,0};
#pragma unroll
    for (int kc = 0; kc < E_; kc += 32) {
        v16h af = load_a_frag(Xh, E_, mBase, kc, lane);
        v16h bf = load_b_frag(W,  E_, nBase, kc, lane);
        acc = wmma_f16(af, bf, acc);
    }

    const int nr  = lane & 15;
    const int off = (lane >> 4) * 8;
    const int n   = nBase + nr;
    const float bb = bias[n];
    const int bIdx = mBase >> 11;     // row block lies within one batch (S=2048)
    const int h    = n >> 5;
    const int d    = n & 31;
    const size_t bh = (size_t)bIdx * H_ + h;

    if (which < 2) {
        _Float16* dst = (which == 0) ? Qh : Kh;
#pragma unroll
        for (int i = 0; i < 8; ++i) {
            int srow = (mBase + off + i) & (S_ - 1);
            dst[(bh * S_ + srow) * HD_ + d] = (_Float16)(acc[i] + bb);
        }
    } else {
#pragma unroll
        for (int i = 0; i < 8; ++i) {
            int srow = (mBase + off + i) & (S_ - 1);
            Vt[(bh * HD_ + d) * S_ + srow] = (_Float16)(acc[i] + bb);
        }
    }
}

// ---------------------------------------------------------------------------
// Pass 1: per-(b,h,k) online max / sum-exp over the QUERY axis (softmax axis=2 quirk).
// Each wave owns 16 key columns and streams all 2048 queries; one WMMA per 16x16 tile
// (K-dim 32 == full head dim). Stores reciprocal of the sum for pass 2.
__global__ __launch_bounds__(128)
void colstats_kernel(const _Float16* __restrict__ Qh,
                     const _Float16* __restrict__ Kh,
                     const float* __restrict__ mask,
                     float* __restrict__ colMax,
                     float* __restrict__ colRcp) {
    const int lane   = threadIdx.x & 31;
    const int w      = blockIdx.x * 4 + (threadIdx.x >> 5);   // 8192 waves
    const int kBlock = w & 127;                               // S/16
    const int h      = (w >> 7) & 7;
    const int b      = w >> 10;
    const int kBase  = kBlock * 16;
    const size_t bh  = (size_t)b * H_ + h;

    const _Float16* Qbh = Qh + bh * (size_t)S_ * HD_;
    const _Float16* Kbh = Kh + bh * (size_t)S_ * HD_;
    const float*    mrw = mask + (size_t)b * S_;
    const int off = (lane >> 4) * 8;

    // B fragment (key tile) is invariant across the q loop.
    const v16h bfrag = load_b_frag(Kbh, HD_, kBase, 0, lane);

    float runM = -1e30f, runL = 0.f;
    for (int q0 = 0; q0 < S_; q0 += 16) {
        v16h af = load_a_frag(Qbh, HD_, q0, 0, lane);
        v8f c = {0,0,0,0,0,0,0,0};
        c = wmma_f16(af, bfrag, c);

        float4 m0 = *(const float4*)(mrw + q0 + off);
        float4 m1 = *(const float4*)(mrw + q0 + off + 4);
        float mr[8] = {m0.x, m0.y, m0.z, m0.w, m1.x, m1.y, m1.z, m1.w};
        float s[8];
#pragma unroll
        for (int i = 0; i < 8; ++i) s[i] = c[i] * INV_SQRT_E + mr[i] * MASK_EPS;

        float tmax = s[0];
#pragma unroll
        for (int i = 1; i < 8; ++i) tmax = fmaxf(tmax, s[i]);
        tmax = fmaxf(tmax, __shfl_xor(tmax, 16, 32));   // combine row halves (lane ^ 16)
        const float newM = fmaxf(runM, tmax);

        float ps = 0.f;
#pragma unroll
        for (int i = 0; i < 8; ++i) ps += __expf(s[i] - newM);
        ps += __shfl_xor(ps, 16, 32);

        runL = runL * __expf(runM - newM) + ps;
        runM = newM;
    }

    if (lane < 16) {
        size_t idx = bh * S_ + kBase + lane;
        colMax[idx] = runM;
        colRcp[idx] = 1.0f / runL;
    }
}

// ---------------------------------------------------------------------------
// Pass 2: recompute scores, normalize per key-column, P @ V into [B,S,E] f16.
// The 4 waves of a block share (b,h) and consecutive q-blocks, so the K / V-transpose
// 32-key chunks are staged once per block into LDS by the Tensor Data Mover, double
// buffered and fenced with s_wait_tensorcnt.
__global__ __launch_bounds__(128)
void attnout_kernel(const _Float16* __restrict__ Qh,
                    const _Float16* __restrict__ Kh,
                    const _Float16* __restrict__ Vt,
                    const float* __restrict__ mask,
                    const float* __restrict__ colMax,
                    const float* __restrict__ colRcp,
                    _Float16* __restrict__ Ah) {
    __shared__ __align__(16) _Float16 Klds[2][32 * HD_];   // [buf][key][d]
    __shared__ __align__(16) _Float16 Vlds[2][HD_ * 32];   // [buf][d][k]
    __shared__ __align__(16) float    P[4][16][36];        // per-wave 16x32 f32 tile (+pad)

    const int lane   = threadIdx.x & 31;
    const int wid    = threadIdx.x >> 5;
    const int w      = blockIdx.x * 4 + wid;               // 8192 waves
    const int qBlock = w & 127;
    const int h      = (w >> 7) & 7;
    const int b      = w >> 10;
    const int qBase  = qBlock * 16;
    const size_t bh  = (size_t)b * H_ + h;

    const _Float16* Qbh = Qh + bh * (size_t)S_ * HD_;
    const _Float16* Kbh = Kh + bh * (size_t)S_ * HD_;
    const _Float16* Vbh = Vt + bh * (size_t)HD_ * S_;
    const float* cmax = colMax + bh * S_;
    const float* crcp = colRcp + bh * S_;
    const int nr  = lane & 15;
    const int off = (lane >> 4) * 8;
    const int hK  = (lane >> 4) * 16;

    // Query A-fragment and mask rows are fixed for this wave.
    const v16h aq = load_a_frag(Qbh, HD_, qBase, 0, lane);
    const float* mrw = mask + (size_t)b * S_ + qBase + off;
    float4 m0 = *(const float4*)(mrw);
    float4 m1 = *(const float4*)(mrw + 4);
    const float mr[8] = {m0.x, m0.y, m0.z, m0.w, m1.x, m1.y, m1.z, m1.w};

    v8f acc0 = {0,0,0,0,0,0,0,0};
    v8f acc1 = {0,0,0,0,0,0,0,0};

    // Prologue: stage first 32-key chunk (K: one contiguous row of 1024 f16;
    // V-transpose: 32 rows x 32 elems, row stride S).
    stage_tile_f16(wid, lane, &Klds[0][0], Kbh, 1024, 1, 1024);
    stage_tile_f16(wid, lane, &Vlds[0][0], Vbh, 32, 32, S_);

    for (int k0 = 0; k0 < S_; k0 += 32) {
        const int buf = (k0 >> 5) & 1;
        if (k0 + 32 < S_) {
            stage_tile_f16(wid, lane, &Klds[buf ^ 1][0], Kbh + (size_t)(k0 + 32) * HD_, 1024, 1, 1024);
            stage_tile_f16(wid, lane, &Vlds[buf ^ 1][0], Vbh + (k0 + 32), 32, 32, S_);
            __builtin_amdgcn_s_wait_tensorcnt(2);   // older pair (current buf) complete
        } else {
            __builtin_amdgcn_s_wait_tensorcnt(0);
        }
        __syncthreads();                            // staged chunk visible to all waves

        // scores for 32 keys = two 16x16 WMMA tiles; normalize per key column (axis=2)
#pragma unroll
        for (int t = 0; t < 2; ++t) {
            v16h bk = *(const v16h*)(&Klds[buf][(16 * t + nr) * HD_ + hK]);
            v8f sc = {0,0,0,0,0,0,0,0};
            sc = wmma_f16(aq, bk, sc);

            const int col = k0 + 16 * t + nr;       // this lane's key column
            const float cm = cmax[col];
            const float rs = crcp[col];
#pragma unroll
            for (int i = 0; i < 8; ++i) {
                float s = sc[i] * INV_SQRT_E + mr[i] * MASK_EPS;
                P[wid][off + i][16 * t + nr] = __expf(s - cm) * rs;
            }
        }
        // V-transpose B-fragments (read before the barrier so buffer reuse is fenced)
        v16h bv0 = *(const v16h*)(&Vlds[buf][(nr)      * 32 + hK]);
        v16h bv1 = *(const v16h*)(&Vlds[buf][(16 + nr) * 32 + hK]);
        __syncthreads();                            // P written

        // Redistribute P (column-per-lane D layout) into A-fragment layout via LDS.
        const float* pr = &P[wid][nr][hK];
        float4 p0 = *(const float4*)(pr);
        float4 p1 = *(const float4*)(pr + 4);
        float4 p2 = *(const float4*)(pr + 8);
        float4 p3 = *(const float4*)(pr + 12);
        const float lo[8] = {p0.x, p0.y, p0.z, p0.w, p1.x, p1.y, p1.z, p1.w};
        const float hi[8] = {p2.x, p2.y, p2.z, p2.w, p3.x, p3.y, p3.z, p3.w};
        v16h ap;
#pragma unroll
        for (int j = 0; j < 8; ++j) {
            ap[j]     = (_Float16)lo[j];
            ap[j + 8] = (_Float16)hi[j];
        }
        __syncthreads();                            // P consumed; K/V reads retired

        acc0 = wmma_f16(ap, bv0, acc0);
        acc1 = wmma_f16(ap, bv1, acc1);
    }

    // Store heads-concatenated [B,S,E] f16: channel = h*32 + d
#pragma unroll
    for (int t = 0; t < 2; ++t) {
        v8f acc = (t == 0) ? acc0 : acc1;
        const int chan = h * HD_ + 16 * t + nr;
#pragma unroll
        for (int i = 0; i < 8; ++i) {
            size_t r = (size_t)b * S_ + qBase + off + i;
            Ah[r * E_ + chan] = (_Float16)acc[i];
        }
    }
}

// ---------------------------------------------------------------------------
// Output projection: out = Ah @ Wo.T + bo, f32 result straight to d_out.
__global__ __launch_bounds__(128)
void out_gemm_kernel(const _Float16* __restrict__ Ah,
                     const _Float16* __restrict__ Woh,
                     const float* __restrict__ bo,
                     float* __restrict__ out) {
    const int lane   = threadIdx.x & 31;
    const int tile   = blockIdx.x * 4 + (threadIdx.x >> 5);
    const int nBlock = tile & 15;
    const int mBlock = tile >> 4;
    const int mBase  = mBlock * 16;
    const int nBase  = nBlock * 16;

    v8f acc = {0,0,0,0,0,0,0,0};
#pragma unroll
    for (int kc = 0; kc < E_; kc += 32) {
        v16h af = load_a_frag(Ah,  E_, mBase, kc, lane);
        v16h bf = load_b_frag(Woh, E_, nBase, kc, lane);
        acc = wmma_f16(af, bf, acc);
    }

    const int nr  = lane & 15;
    const int off = (lane >> 4) * 8;
    const int n   = nBase + nr;
    const float bb = bo[n];
#pragma unroll
    for (int i = 0; i < 8; ++i)
        out[(size_t)(mBase + off + i) * E_ + n] = acc[i] + bb;
}

// ---------------------------------------------------------------------------
extern "C" void kernel_launch(void* const* d_in, const int* in_sizes, int n_in,
                              void* d_out, int out_size, void* d_ws, size_t ws_size,
                              hipStream_t stream) {
    (void)in_sizes; (void)n_in; (void)out_size; (void)ws_size;

    const float* X    = (const float*)d_in[0];
    const float* mask = (const float*)d_in[1];
    const float* Wq   = (const float*)d_in[2]; const float* bq = (const float*)d_in[3];
    const float* Wk   = (const float*)d_in[4]; const float* bk = (const float*)d_in[5];
    const float* Wv   = (const float*)d_in[6]; const float* bv = (const float*)d_in[7];
    const float* Wo   = (const float*)d_in[8]; const float* bo = (const float*)d_in[9];
    float* out = (float*)d_out;

    char* ws = (char*)d_ws;
    size_t o = 0;
    auto carve = [&](size_t bytes) -> void* {
        void* p = ws + o;
        o += (bytes + 255) & ~(size_t)255;
        return p;
    };
    _Float16* Xh  = (_Float16*)carve((size_t)BS_ * E_ * 2);
    _Float16* Qh  = (_Float16*)carve((size_t)BS_ * E_ * 2);
    _Float16* Kh  = (_Float16*)carve((size_t)BS_ * E_ * 2);
    _Float16* Vt  = (_Float16*)carve((size_t)BS_ * E_ * 2);
    _Float16* Ah  = (_Float16*)carve((size_t)BS_ * E_ * 2);
    _Float16* Wqh = (_Float16*)carve((size_t)E_ * E_ * 2);
    _Float16* Wkh = (_Float16*)carve((size_t)E_ * E_ * 2);
    _Float16* Wvh = (_Float16*)carve((size_t)E_ * E_ * 2);
    _Float16* Woh = (_Float16*)carve((size_t)E_ * E_ * 2);
    float* colMax = (float*)carve((size_t)B_ * H_ * S_ * 4);
    float* colRcp = (float*)carve((size_t)B_ * H_ * S_ * 4);

    // f32 -> f16 staging
    cvt_f32_to_f16<<<(BS_ * E_ + 255) / 256, 256, 0, stream>>>(X,  Xh,  BS_ * E_);
    cvt_f32_to_f16<<<(E_ * E_ + 255) / 256, 256, 0, stream>>>(Wq, Wqh, E_ * E_);
    cvt_f32_to_f16<<<(E_ * E_ + 255) / 256, 256, 0, stream>>>(Wk, Wkh, E_ * E_);
    cvt_f32_to_f16<<<(E_ * E_ + 255) / 256, 256, 0, stream>>>(Wv, Wvh, E_ * E_);
    cvt_f32_to_f16<<<(E_ * E_ + 255) / 256, 256, 0, stream>>>(Wo, Woh, E_ * E_);

    // Q/K/V projections (16384 wave-tiles each, fused over blockIdx.y)
    qkv_gemm_kernel<<<dim3(4096, 3), 128, 0, stream>>>(Xh, Wqh, Wkh, Wvh,
                                                       bq, bk, bv, Qh, Kh, Vt);
    // Softmax-over-query column statistics (8192 waves)
    colstats_kernel<<<2048, 128, 0, stream>>>(Qh, Kh, mask, colMax, colRcp);
    // Normalized P @ V with TDM-staged, double-buffered K/V chunks (8192 waves)
    attnout_kernel<<<2048, 128, 0, stream>>>(Qh, Kh, Vt, mask, colMax, colRcp, Ah);
    // Output projection
    out_gemm_kernel<<<4096, 128, 0, stream>>>(Ah, Woh, bo, out);
}
